// Sarnet_5205500363305
// MI455X (gfx1250) — matrix operand order repaired and checked by reading, not compiled, not run on hardware
//
#include <hip/hip_runtime.h>
#include <hip/hip_bf16.h>

typedef _Float16 v8h  __attribute__((ext_vector_type(8)));
typedef _Float16 v16h __attribute__((ext_vector_type(16)));
typedef float    v8f  __attribute__((ext_vector_type(8)));

#define NB      16384
#define NF      32
#define NE      16
#define NV      100000
#define NI      512
#define ND      4
#define NS      8
#define NK      2
#define NOUT    16
#define N1      144   // 128 shared-expert cols + 10 gate cols + 6 pad
#define N2      128   // 4 domains * 2 experts * 16
#define BNEPS   1e-5f

// ---------------- workspace layout (bytes) ----------------
#define OFF_EMB16   ((size_t)0)                          // [B,512] f16
#define OFF_SH16    (OFF_EMB16 + (size_t)NB*NI*2)        // [B,512] f16
#define OFF_OUT1    (OFF_SH16  + (size_t)NB*NI*2)        // [B,144] f32
#define OFF_OUT2    (OFF_OUT1  + (size_t)NB*N1*4)        // [B,128] f32
#define OFF_WCT1    (OFF_OUT2  + (size_t)NB*N2*4)        // [144,512] f16 (transposed)
#define OFF_WCT2    (OFF_WCT1  + (size_t)N1*NI*2)        // [128,512] f16 (transposed)
#define OFF_BIAS1   (OFF_WCT2  + (size_t)N2*NI*2)        // [144] f32 (pad to 1KB)
#define OFF_BIAS2   (OFF_BIAS1 + (size_t)1024)           // [128] f32 (pad to 1KB)
#define OFF_STATS   (OFF_BIAS2 + (size_t)1024)           // [4*512] f32: sum_e,sq_e,sum_s,sq_s
#define OFF_H1PRE   (OFF_STATS + (size_t)4*NI*4)         // [B,32] f32
#define OFF_H2PRE   (OFF_H1PRE + (size_t)NB*32*4)        // [B,32] f32
#define OFF_ST1     (OFF_H2PRE + (size_t)NB*32*4)        // [64] f32
#define OFF_ST2     (OFF_ST1   + (size_t)256)            // [64] f32

// ---------------- 1) embedding gather + per-domain affine, f16 store --------
__global__ __launch_bounds__(256) void k_gather(
    const int* __restrict__ ids, const int* __restrict__ dom,
    const float* __restrict__ table,
    const float* __restrict__ dw, const float* __restrict__ db,
    _Float16* __restrict__ emb16, _Float16* __restrict__ sh16)
{
    int t = threadIdx.x;
    int f = t & 31;                 // field (F == 32)
    int r = t >> 5;                 // 8 rows per block
    int b = blockIdx.x * 8 + r;
    int id = ids[b * NF + f];
    int d  = dom[b];
    const float* src = table + ((size_t)f * NV + id) * NE;
    const float* dwp = dw + d * NI + f * NE;
    const float* dbp = db + d * NI + f * NE;
    v8h e0, e1, s0, s1;
#pragma unroll
    for (int e = 0; e < 8; ++e) {
        float v = src[e];
        e0[e] = (_Float16)v;
        s0[e] = (_Float16)(v * dwp[e] + dbp[e]);
    }
#pragma unroll
    for (int e = 0; e < 8; ++e) {
        float v = src[8 + e];
        e1[e] = (_Float16)v;
        s1[e] = (_Float16)(v * dwp[8 + e] + dbp[8 + e]);
    }
    size_t o = (size_t)b * NI + f * NE;
    *(v8h*)(emb16 + o)     = e0;
    *(v8h*)(emb16 + o + 8) = e1;
    *(v8h*)(sh16 + o)      = s0;
    *(v8h*)(sh16 + o + 8)  = s1;
}

// ---------------- 2) per-feature batch sums for emb and shared_emb ----------
__global__ __launch_bounds__(256) void k_stats512(
    const _Float16* __restrict__ emb16, const _Float16* __restrict__ sh16,
    float* __restrict__ stats)
{
    int arr = blockIdx.x >> 4;               // 0: emb, 1: shared_emb
    int f0  = (blockIdx.x & 15) * 32;
    const _Float16* src = arr ? sh16 : emb16;
    int lane = threadIdx.x & 31, r = threadIdx.x >> 5;
    int f = f0 + lane;
    float s = 0.f, s2 = 0.f;
    for (int b = r; b < NB; b += 8) {
        float v = (float)src[(size_t)b * NI + f];
        s += v; s2 += v * v;
    }
    __shared__ float ls[8][32], ls2[8][32];
    ls[r][lane] = s; ls2[r][lane] = s2;
    __syncthreads();
    if (r == 0) {
        float ts = 0.f, ts2 = 0.f;
        for (int rr = 0; rr < 8; ++rr) { ts += ls[rr][lane]; ts2 += ls2[rr][lane]; }
        stats[arr * 1024 + f]       = ts;
        stats[arr * 1024 + 512 + f] = ts2;
    }
}

// ---------------- 3) fold BN + gamma into weight matrices (transposed) ------
__global__ __launch_bounds__(256) void k_prep(
    const float* __restrict__ stats, const float* __restrict__ dw,
    const float* __restrict__ sh_gamma, const float* __restrict__ sh_beta,
    const float* __restrict__ sh_W, const float* __restrict__ sh_b,
    const float* __restrict__ sp_gamma, const float* __restrict__ sp_beta,
    const float* __restrict__ sp_W, const float* __restrict__ sp_b,
    const float* __restrict__ gate_W, const float* __restrict__ gate_b,
    _Float16* __restrict__ wct1, _Float16* __restrict__ wct2,
    float* __restrict__ bias1, float* __restrict__ bias2)
{
    int n = blockIdx.x;                  // 0..143: GEMM1 col; 144..271: GEMM2 col
    int t = threadIdx.x;
    const float invB = 1.0f / (float)NB;
    float bacc = 0.f;
    if (n < N1) {
        if (n < 128) {                   // shared experts, BN folded
            int s = n >> 4, o = n & 15;
            for (int i = t; i < NI; i += 256) {
                float msh = stats[1024 + i] * invB;
                float vsh = stats[1536 + i] * invB - msh * msh;
                float inv = rsqrtf(vsh + BNEPS);
                float g = sh_gamma[s * NI + i];
                float w = sh_W[((size_t)(s * NI + i)) * NOUT + o];
                wct1[(size_t)n * NI + i] = (_Float16)(inv * g * w);
                bacc += (sh_beta[s * NI + i] - msh * inv * g) * w;
            }
        } else if (n < 138) {            // gate columns (raw shared_emb, no BN)
            int g = n - 128;
            for (int i = t; i < NI; i += 256)
                wct1[(size_t)n * NI + i] = (_Float16)gate_W[i * (NS + NK) + g];
        } else {                         // pad
            for (int i = t; i < NI; i += 256)
                wct1[(size_t)n * NI + i] = (_Float16)0.0f;
        }
    } else {                             // domain-specific experts, analytic BN
        int n2 = n - N1;
        int d = n2 >> 5, k = (n2 >> 4) & 1, o = n2 & 15;
        int dk = d * NK + k;
        for (int i = t; i < NI; i += 256) {
            float me  = stats[i] * invB;
            float ve  = stats[512 + i] * invB - me * me;
            float dwv = dw[d * NI + i];
            float sd  = dwv * rsqrtf(ve * dwv * dwv + BNEPS);
            float g = sp_gamma[dk * NI + i];
            float w = sp_W[((size_t)(dk * NI + i)) * NOUT + o];
            wct2[(size_t)n2 * NI + i] = (_Float16)(sd * g * w);
            bacc += (sp_beta[dk * NI + i] - me * sd * g) * w;
        }
    }
    __shared__ float red[256];
    red[t] = bacc;
    __syncthreads();
    for (int off = 128; off > 0; off >>= 1) {
        if (t < off) red[t] += red[t + off];
        __syncthreads();
    }
    if (t == 0) {
        if (n < 128)      { int s = n >> 4, o = n & 15; bias1[n] = red[0] + sh_b[s * NOUT + o]; }
        else if (n < 138) bias1[n] = gate_b[n - 128];
        else if (n < N1)  bias1[n] = 0.f;
        else { int n2 = n - N1; int d = n2 >> 5, k = (n2 >> 4) & 1, o = n2 & 15;
               bias2[n2] = red[0] + sp_b[(d * NK + k) * NOUT + o]; }
    }
}

// ---------------- 4) dual GEMM via v_wmma_f32_16x16x32_f16 ------------------
// 17 column tiles total: 9 for GEMM1 (shared_emb x Wc1), 8 for GEMM2 (emb x Wc2).
// One wave per 16x16 tile, K=512 in 16 WMMA steps.
__global__ __launch_bounds__(256) void k_gemm(
    const _Float16* __restrict__ emb16, const _Float16* __restrict__ sh16,
    const _Float16* __restrict__ wct1, const _Float16* __restrict__ wct2,
    const float* __restrict__ bias1, const float* __restrict__ bias2,
    float* __restrict__ out1, float* __restrict__ out2)
{
    int wave = blockIdx.x * 8 + (threadIdx.x >> 5);
    int lane = threadIdx.x & 31;
    int rt = wave / 17;
    int ct = wave % 17;
    const _Float16* A; const _Float16* BT; const float* bias;
    float* outp; int ldOut, n0;
    if (ct < 9) { A = sh16;  BT = wct1; bias = bias1; outp = out1; ldOut = N1; n0 = ct * 16; }
    else        { A = emb16; BT = wct2; bias = bias2; outp = out2; ldOut = N2; n0 = (ct - 9) * 16; }
    int m = lane & 15, h = lane >> 4;
    const _Float16* arow = A  + (size_t)(rt * 16 + m) * NI;
    const _Float16* brow = BT + (size_t)(n0 + m) * NI;      // transposed weights
    v8f acc = {0.f, 0.f, 0.f, 0.f, 0.f, 0.f, 0.f, 0.f};
#pragma unroll
    for (int kk = 0; kk < 16; ++kk) {
        int k0 = kk * 32;
        // A 16x32 f16 lane layout: lane half h holds K in [8h,8h+8) U [16+8h,16+8h+8)
        v8h a0 = *(const v8h*)(arow + k0 + 8 * h);
        v8h a1 = *(const v8h*)(arow + k0 + 16 + 8 * h);
        // B 32x16 f16 lane layout: lane n=lane%16 holds K in [8h,8h+8) U [16+8h,16+8h+8)
        v8h b0 = *(const v8h*)(brow + k0 + 8 * h);
        v8h b1 = *(const v8h*)(brow + k0 + 16 + 8 * h);
        v16h a = __builtin_shufflevector(a0, a1, 0,1,2,3,4,5,6,7,8,9,10,11,12,13,14,15);
        v16h bb = __builtin_shufflevector(b0, b1, 0,1,2,3,4,5,6,7,8,9,10,11,12,13,14,15);
        acc = __builtin_amdgcn_wmma_f32_16x16x32_f16(false, a, false, bb,
                                                     (short)0, acc, false, false);
    }
    float bv = bias[n0 + m];
#pragma unroll
    for (int r = 0; r < 8; ++r) {                 // D layout: VGPR r -> row r+8h
        int row = rt * 16 + r + 8 * h;
        outp[(size_t)row * ldOut + n0 + m] = acc[r] + bv;
    }
}

// ---------------- 5) softmax gate + expert mix + MLP layer 1 ----------------
__global__ __launch_bounds__(256) void k_mix(
    const float* __restrict__ out1, const float* __restrict__ out2,
    const int* __restrict__ dom,
    const float* __restrict__ W1, const float* __restrict__ b1,
    float* __restrict__ h1pre)
{
    int lane = threadIdx.x & 31, r = threadIdx.x >> 5;
    int b = blockIdx.x * 8 + r;
    const float* l = out1 + (size_t)b * N1 + 128;   // 10 gate logits
    float g[10];
    float mx = l[0];
#pragma unroll
    for (int e = 1; e < 10; ++e) mx = fmaxf(mx, l[e]);
    float sum = 0.f;
#pragma unroll
    for (int e = 0; e < 10; ++e) { g[e] = __expf(l[e] - mx); sum += g[e]; }
    float inv = 1.0f / sum;
    __shared__ float mix[8][16];
    if (lane < 16) {
        float acc = 0.f;
        const float* o1 = out1 + (size_t)b * N1;
#pragma unroll
        for (int s = 0; s < 8; ++s) acc += g[s] * o1[s * 16 + lane];
        int d = dom[b];
        const float* o2 = out2 + (size_t)b * N2 + d * 32;
        acc += g[8] * o2[lane] + g[9] * o2[16 + lane];
        mix[r][lane] = acc * inv;
    }
    __syncthreads();
    float hv = b1[lane];
#pragma unroll
    for (int o = 0; o < 16; ++o) hv += mix[r][o] * W1[o * 32 + lane];
    h1pre[(size_t)b * 32 + lane] = hv;
}

// ---------------- 6) 32-feature batch stats (deterministic) -----------------
__global__ __launch_bounds__(256) void k_stats32(
    const float* __restrict__ src, float* __restrict__ st)
{
    int lane = threadIdx.x & 31, r = threadIdx.x >> 5;
    float s = 0.f, s2 = 0.f;
    for (int b = r; b < NB; b += 8) {
        float v = src[(size_t)b * 32 + lane];
        s += v; s2 += v * v;
    }
    __shared__ float ls[8][32], ls2[8][32];
    ls[r][lane] = s; ls2[r][lane] = s2;
    __syncthreads();
    if (r == 0) {
        float ts = 0.f, ts2 = 0.f;
        for (int rr = 0; rr < 8; ++rr) { ts += ls[rr][lane]; ts2 += ls2[rr][lane]; }
        st[lane] = ts; st[32 + lane] = ts2;
    }
}

// ---------------- 7) BN1 + ReLU + MLP layer 2 -------------------------------
__global__ __launch_bounds__(256) void k_mlp2(
    const float* __restrict__ h1pre, const float* __restrict__ st1,
    const float* __restrict__ g1, const float* __restrict__ be1,
    const float* __restrict__ W2, const float* __restrict__ b2,
    float* __restrict__ h2pre)
{
    int lane = threadIdx.x & 31, r = threadIdx.x >> 5;
    int b = blockIdx.x * 8 + r;
    const float invB = 1.0f / (float)NB;
    float m = st1[lane] * invB;
    float v = st1[32 + lane] * invB - m * m;
    float inv = rsqrtf(v + BNEPS);
    float h = (h1pre[(size_t)b * 32 + lane] - m) * inv * g1[lane] + be1[lane];
    h = fmaxf(h, 0.f);
    __shared__ float hs[8][32];
    hs[r][lane] = h;
    __syncthreads();
    float acc = b2[lane];
#pragma unroll
    for (int o = 0; o < 32; ++o) acc += hs[r][o] * W2[o * 32 + lane];
    h2pre[(size_t)b * 32 + lane] = acc;
}

// ---------------- 8) BN2 + ReLU + MLP layer 3 + sigmoid ---------------------
__global__ __launch_bounds__(256) void k_final(
    const float* __restrict__ h2pre, const float* __restrict__ st2,
    const float* __restrict__ g2, const float* __restrict__ be2,
    const float* __restrict__ W3, const float* __restrict__ b3,
    float* __restrict__ out)
{
    int lane = threadIdx.x & 31, r = threadIdx.x >> 5;
    int b = blockIdx.x * 8 + r;
    const float invB = 1.0f / (float)NB;
    float m = st2[lane] * invB;
    float v = st2[32 + lane] * invB - m * m;
    float inv = rsqrtf(v + BNEPS);
    float h = (h2pre[(size_t)b * 32 + lane] - m) * inv * g2[lane] + be2[lane];
    h = fmaxf(h, 0.f);
    float p = h * W3[lane];
#pragma unroll
    for (int off = 16; off > 0; off >>= 1) p += __shfl_xor(p, off, 32);
    if (lane == 0) out[b] = 1.0f / (1.0f + __expf(-(p + b3[0])));
}

extern "C" void kernel_launch(void* const* d_in, const int* in_sizes, int n_in,
                              void* d_out, int out_size, void* d_ws, size_t ws_size,
                              hipStream_t stream) {
    const int*   ids       = (const int*)d_in[0];
    const int*   domain_id = (const int*)d_in[1];
    const float* emb_table = (const float*)d_in[2];
    const float* dw        = (const float*)d_in[3];
    const float* db        = (const float*)d_in[4];
    const float* sh_gamma  = (const float*)d_in[5];
    const float* sh_beta   = (const float*)d_in[6];
    const float* sh_W      = (const float*)d_in[7];
    const float* sh_b      = (const float*)d_in[8];
    const float* sp_gamma  = (const float*)d_in[9];
    const float* sp_beta   = (const float*)d_in[10];
    const float* sp_W      = (const float*)d_in[11];
    const float* sp_b      = (const float*)d_in[12];
    const float* gate_W    = (const float*)d_in[13];
    const float* gate_b    = (const float*)d_in[14];
    const float* mlp_W1    = (const float*)d_in[15];
    const float* mlp_b1    = (const float*)d_in[16];
    const float* mlp_g1    = (const float*)d_in[17];
    const float* mlp_be1   = (const float*)d_in[18];
    const float* mlp_W2    = (const float*)d_in[19];
    const float* mlp_b2    = (const float*)d_in[20];
    const float* mlp_g2    = (const float*)d_in[21];
    const float* mlp_be2   = (const float*)d_in[22];
    const float* mlp_W3    = (const float*)d_in[23];
    const float* mlp_b3    = (const float*)d_in[24];

    char* ws = (char*)d_ws;
    _Float16* emb16 = (_Float16*)(ws + OFF_EMB16);
    _Float16* sh16  = (_Float16*)(ws + OFF_SH16);
    float*    out1  = (float*)(ws + OFF_OUT1);
    float*    out2  = (float*)(ws + OFF_OUT2);
    _Float16* wct1  = (_Float16*)(ws + OFF_WCT1);
    _Float16* wct2  = (_Float16*)(ws + OFF_WCT2);
    float*    bias1 = (float*)(ws + OFF_BIAS1);
    float*    bias2 = (float*)(ws + OFF_BIAS2);
    float*    stats = (float*)(ws + OFF_STATS);
    float*    h1pre = (float*)(ws + OFF_H1PRE);
    float*    h2pre = (float*)(ws + OFF_H2PRE);
    float*    st1   = (float*)(ws + OFF_ST1);
    float*    st2   = (float*)(ws + OFF_ST2);
    float*    outp  = (float*)d_out;

    k_gather<<<NB / 8, 256, 0, stream>>>(ids, domain_id, emb_table, dw, db, emb16, sh16);
    k_stats512<<<32, 256, 0, stream>>>(emb16, sh16, stats);
    k_prep<<<N1 + N2, 256, 0, stream>>>(stats, dw, sh_gamma, sh_beta, sh_W, sh_b,
                                        sp_gamma, sp_beta, sp_W, sp_b, gate_W, gate_b,
                                        wct1, wct2, bias1, bias2);
    // (B/16 row tiles) * 17 col tiles = 17408 waves, 8 waves per block
    k_gemm<<<(NB / 16) * 17 / 8, 256, 0, stream>>>(emb16, sh16, wct1, wct2,
                                                   bias1, bias2, out1, out2);
    k_mix<<<NB / 8, 256, 0, stream>>>(out1, out2, domain_id, mlp_W1, mlp_b1, h1pre);
    k_stats32<<<1, 256, 0, stream>>>(h1pre, st1);
    k_mlp2<<<NB / 8, 256, 0, stream>>>(h1pre, st1, mlp_g1, mlp_be1, mlp_W2, mlp_b2, h2pre);
    k_stats32<<<1, 256, 0, stream>>>(h2pre, st2);
    k_final<<<NB / 8, 256, 0, stream>>>(h2pre, st2, mlp_g2, mlp_be2, mlp_W3, mlp_b3, outp);
}
